// GemmaAttention_78554951843967
// MI455X (gfx1250) — compile-verified, matrix operand order checked by reading, not run on hardware
//
#include <hip/hip_runtime.h>
#include <hip/hip_bf16.h>

// ---------------------------------------------------------------------------
// Gemma-style attention for gfx1250 (MI455X), wave32 + v_wmma_f32_16x16x32_f16
// + async global->LDS staging (ASYNCcnt) in the GEMMs.
// ---------------------------------------------------------------------------

typedef __attribute__((ext_vector_type(16))) _Float16 v16h;
typedef __attribute__((ext_vector_type(8)))  _Float16 v8h;
typedef __attribute__((ext_vector_type(4)))  _Float16 v4h;
typedef __attribute__((ext_vector_type(8)))  float    v8f;
typedef int v4i_ __attribute__((vector_size(16)));

#define B_   2
#define S_   2048
#define DIM_ 2048
#define H_   16
#define KV_  8
#define D_   128
#define QKVC 4096          // (H + 2*KV) * D
#define SCALE_ 0.08838834764831845f   // D^-0.5
#define NEG_ -1000000000.0f

__device__ __forceinline__ v8f wmma_f16(v16h a, v16h b, v8f c) {
    // D = A(16x32 f16) * B(32x16 f16) + C(16x16 f32)
    return __builtin_amdgcn_wmma_f32_16x16x32_f16(
        false, a, false, b, (short)0, c, false, false);
}

// 16x32 f16 fragment (A layout; identical striping used for B):
//   lane: m = lane&15, khalf = lane>>4
//   elems 0..7  -> K = khalf*8 + e        (contiguous 16B)
//   elems 8..15 -> K = 16 + khalf*8 + e   (contiguous 16B)
__device__ __forceinline__ v16h load_frag(const _Float16* __restrict__ base,
                                          int ld, int row, int k0) {
    const int khalf = (threadIdx.x >> 4) & 1;
    const _Float16* p = base + (size_t)row * ld + k0 + khalf * 8;
    v8h lo = *(const v8h*)p;
    v8h hi = *(const v8h*)(p + 16);
    v16h r;
#pragma unroll
    for (int i = 0; i < 8; ++i) { r[i] = lo[i]; r[i + 8] = hi[i]; }
    return r;
}

// Same fragment shape out of an LDS tile with leading dim 32 halves.
__device__ __forceinline__ v16h lds_frag(const _Float16* base, int row) {
    const int khalf = (threadIdx.x >> 4) & 1;
    const _Float16* p = base + row * 32 + khalf * 8;
    v8h lo = *(const v8h*)p;
    v8h hi = *(const v8h*)(p + 16);
    v16h r;
#pragma unroll
    for (int i = 0; i < 8; ++i) { r[i] = lo[i]; r[i + 8] = hi[i]; }
    return r;
}

// ---- async global->LDS copy (16B per lane), with graceful fallback --------
#if __has_builtin(__builtin_amdgcn_global_load_async_to_lds_b128)
#define HAVE_ASYNC_LDS 1
__device__ __forceinline__ void cp_b128(const _Float16* g, _Float16* l) {
    __builtin_amdgcn_global_load_async_to_lds_b128(
        (__attribute__((address_space(1))) v4i_*)g,
        (__attribute__((address_space(3))) v4i_*)l, 0, 0);
}
#else
#define HAVE_ASYNC_LDS 0
__device__ __forceinline__ void cp_b128(const _Float16* g, _Float16* l) {
    *(v8h*)l = *(const v8h*)g;
}
#endif

__device__ __forceinline__ void wait_async0() {
#if HAVE_ASYNC_LDS
#if __has_builtin(__builtin_amdgcn_s_wait_asynccnt)
    __builtin_amdgcn_s_wait_asynccnt(0);
#else
    asm volatile("s_wait_asynccnt 0" ::: "memory");
#endif
#endif
}

// ---------------------------------------------------------------------------
// Kernel 1: fp32 -> fp16 conversion, 4-wide
// ---------------------------------------------------------------------------
__global__ void cvt_f32_f16(const float4* __restrict__ in,
                            v4h* __restrict__ out, int n4) {
    int i = blockIdx.x * blockDim.x + threadIdx.x;
    if (i < n4) {
        float4 x = in[i];
        v4h h;
        h[0] = (_Float16)x.x; h[1] = (_Float16)x.y;
        h[2] = (_Float16)x.z; h[3] = (_Float16)x.w;
        out[i] = h;
    }
}

// ---------------------------------------------------------------------------
// Kernel 2/5: C[M][N] (f32) = A[M][K] (f16 row-major) * Bw[N][K]^T (f16)
// Block: 128 threads = 4 waves (2x2); block tile 64(M) x 128(N); k-step 32.
// Double-buffered LDS staging via async global->LDS (24 KB / block).
// N, K compile-time so the store offsets fold into immediates.
// Grid: (N/128, M/64).
// ---------------------------------------------------------------------------
template<int N, int K>
__global__ __launch_bounds__(128)
void gemm_f16_wmma(const _Float16* __restrict__ A,
                   const _Float16* __restrict__ Bw,
                   float* __restrict__ C) {
    __shared__ _Float16 sA[2][64 * 32];    // 4 KB per buffer
    __shared__ _Float16 sB[2][128 * 32];   // 8 KB per buffer

    const int tid  = threadIdx.x;
    const int lane = tid & 31;
    const int wv   = tid >> 5;
    const int wm   = wv >> 1;               // wave row 0..1 (32 M each)
    const int wn   = wv & 1;                // wave col 0..1 (64 N each)
    const int row0 = blockIdx.y * 64;
    const int col0 = blockIdx.x * 128;

    const _Float16* Ag = A  + (size_t)row0 * K;
    const _Float16* Bg = Bw + (size_t)col0 * K;

    // Cooperative stage of one k-slice (A: 64x32, B: 128x32) into buffer `bi`.
    auto stage = [&](int k0, int bi) {
#pragma unroll
        for (int c = 0; c < 2; ++c) {       // A tile: 256 x 16B chunks
            int ch = tid + c * 128;
            int r = ch >> 2, cc = (ch & 3) * 8;
            cp_b128(Ag + (size_t)r * K + k0 + cc, &sA[bi][r * 32 + cc]);
        }
#pragma unroll
        for (int c = 0; c < 4; ++c) {       // B tile: 512 x 16B chunks
            int ch = tid + c * 128;
            int r = ch >> 2, cc = (ch & 3) * 8;
            cp_b128(Bg + (size_t)r * K + k0 + cc, &sB[bi][r * 32 + cc]);
        }
    };

    v8f acc[2][4];
#pragma unroll
    for (int i = 0; i < 2; ++i)
#pragma unroll
        for (int j = 0; j < 4; ++j) acc[i][j] = {};

    const int m15 = lane & 15;

    stage(0, 0);
    wait_async0();
    __syncthreads();

    int buf = 0;
#pragma unroll 1
    for (int k0 = 0; k0 < K; k0 += 32) {
        const bool more = (k0 + 32) < K;
        if (more) stage(k0 + 32, buf ^ 1);

        v16h a0 = lds_frag(&sA[buf][0], wm * 32 + m15);
        v16h a1 = lds_frag(&sA[buf][0], wm * 32 + 16 + m15);
#pragma unroll
        for (int j = 0; j < 4; ++j) {
            v16h bf = lds_frag(&sB[buf][0], wn * 64 + j * 16 + m15);
            acc[0][j] = wmma_f16(a0, bf, acc[0][j]);
            acc[1][j] = wmma_f16(a1, bf, acc[1][j]);
        }

        if (more) wait_async0();
        __syncthreads();
        buf ^= 1;
    }

    // Epilogue: constant-offset stores (N is compile-time).
    const int khalf = (lane >> 4) & 1;
    float* cp = C + (size_t)(row0 + wm * 32 + khalf * 8) * N
                  + col0 + wn * 64 + m15;
#pragma unroll
    for (int i = 0; i < 2; ++i)
#pragma unroll
        for (int j = 0; j < 4; ++j)
#pragma unroll
            for (int v = 0; v < 8; ++v)
                cp[(size_t)(i * 16 + v) * N + j * 16] = acc[i][j][v];
}

// ---------------------------------------------------------------------------
// Kernel 3: fused RMSNorm + RoPE + layout shuffle + f16 convert.
// One wave per (b, s, head32) vector of 128; lane owns d = {l, l+32, l+64, l+96}.
// Writes: Qh [b][h][s][d], Kh [b][kv][s][d], Vt [b][kv][d][s] (transposed).
// ---------------------------------------------------------------------------
__global__ __launch_bounds__(256)
void normrope_kernel(const float* __restrict__ qkv,
                     const float* __restrict__ qw, const float* __restrict__ kw,
                     const float* __restrict__ cosT, const float* __restrict__ sinT,
                     _Float16* __restrict__ Qh, _Float16* __restrict__ Kh,
                     _Float16* __restrict__ Vt) {
    const int lane = threadIdx.x & 31;
    const int wgid = blockIdx.x * (blockDim.x >> 5) + (threadIdx.x >> 5);
    const int head = wgid & 31;          // 0..15 q, 16..23 k, 24..31 v
    const int bs   = wgid >> 5;          // b*S + s
    const int s    = bs & (S_ - 1);
    const int b    = bs >> 11;

    const float* row = qkv + (size_t)bs * QKVC + head * D_;
    float x0 = row[lane];
    float x1 = row[lane + 32];
    float x2 = row[lane + 64];
    float x3 = row[lane + 96];

    if (head < 24) {
        float ssum = x0 * x0 + x1 * x1 + x2 * x2 + x3 * x3;
#pragma unroll
        for (int off = 16; off >= 1; off >>= 1)
            ssum += __shfl_xor(ssum, off, 32);
        float r = rsqrtf(ssum * (1.0f / 128.0f) + 1e-6f);
        const float* w = (head < 16) ? qw : kw;
        x0 = x0 * r * (1.0f + w[lane]);
        x1 = x1 * r * (1.0f + w[lane + 32]);
        x2 = x2 * r * (1.0f + w[lane + 64]);
        x3 = x3 * r * (1.0f + w[lane + 96]);
        float c0 = cosT[s * 64 + lane],      sn0 = sinT[s * 64 + lane];
        float c1 = cosT[s * 64 + lane + 32], sn1 = sinT[s * 64 + lane + 32];
        float o0 = x0 * c0 - x2 * sn0;
        float o2 = x0 * sn0 + x2 * c0;
        float o1 = x1 * c1 - x3 * sn1;
        float o3 = x1 * sn1 + x3 * c1;
        _Float16* dst;
        if (head < 16)
            dst = Qh + ((size_t)(b * H_ + head) * S_ + s) * D_;
        else
            dst = Kh + ((size_t)(b * KV_ + (head - 16)) * S_ + s) * D_;
        dst[lane]      = (_Float16)o0;
        dst[lane + 32] = (_Float16)o1;
        dst[lane + 64] = (_Float16)o2;
        dst[lane + 96] = (_Float16)o3;
    } else {
        const int kv = head - 24;
        _Float16* dst = Vt + (size_t)(b * KV_ + kv) * D_ * S_;
        dst[(size_t)(lane)      * S_ + s] = (_Float16)x0;
        dst[(size_t)(lane + 32) * S_ + s] = (_Float16)x1;
        dst[(size_t)(lane + 64) * S_ + s] = (_Float16)x2;
        dst[(size_t)(lane + 96) * S_ + s] = (_Float16)x3;
    }
}

// ---------------------------------------------------------------------------
// Kernel 4: flash attention. One wave per 16-row Q tile; 4 waves per block.
// 32-key blocks: scores via WMMA, causal mask, online softmax,
// P restriped through LDS (C-layout -> A-layout), O accumulated via WMMA.
// ---------------------------------------------------------------------------
__global__ __launch_bounds__(128)
void attn_kernel(const _Float16* __restrict__ Qh,
                 const _Float16* __restrict__ Kh,
                 const _Float16* __restrict__ Vt,
                 _Float16* __restrict__ AO) {
    __shared__ _Float16 ldsP[4][16 * 32];   // 1KB P tile per wave

    const int lane  = threadIdx.x & 31;
    const int wv    = threadIdx.x >> 5;
    const int khalf = (lane >> 4) & 1;
    const int n     = lane & 15;

    const int tile = blockIdx.x * 4 + wv;   // B*H*(S/16) = 4096 tiles
    const int qt   = tile & 127;
    const int h    = (tile >> 7) & 15;
    const int b    = tile >> 11;
    const int q0   = qt * 16;
    const int kvh  = h >> 1;                // GQA: H/KV = 2

    const _Float16* Qbase = Qh + (size_t)(b * H_ + h) * S_ * D_;
    const _Float16* Kbase = Kh + (size_t)(b * KV_ + kvh) * S_ * D_;
    const _Float16* Vbase = Vt + (size_t)(b * KV_ + kvh) * D_ * S_;

    v16h qf[4];
#pragma unroll
    for (int kc = 0; kc < 4; ++kc)
        qf[kc] = load_frag(Qbase, D_, q0 + n, kc * 32);

    v8f acc[8];
#pragma unroll
    for (int t = 0; t < 8; ++t) acc[t] = {};

    float mrun[8], lrun[8];
#pragma unroll
    for (int v = 0; v < 8; ++v) { mrun[v] = -3.0e38f; lrun[v] = 0.0f; }

    const int rbase = q0 + khalf * 8;

#pragma unroll 1
    for (int kb = 0; kb < q0 + 16; kb += 32) {
        v8f s0 = {}, s1 = {};
#pragma unroll
        for (int kc = 0; kc < 4; ++kc) {
            v16h b0 = load_frag(Kbase, D_, kb + n, kc * 32);
            v16h b1 = load_frag(Kbase, D_, kb + 16 + n, kc * 32);
            s0 = wmma_f16(qf[kc], b0, s0);
            s1 = wmma_f16(qf[kc], b1, s1);
        }

        float p0[8], p1[8];
#pragma unroll
        for (int v = 0; v < 8; ++v) {
            const int row = rbase + v;
            p0[v] = s0[v] * SCALE_ + ((kb + n)      > row ? NEG_ : 0.0f);
            p1[v] = s1[v] * SCALE_ + ((kb + 16 + n) > row ? NEG_ : 0.0f);
        }

        float mx[8];
#pragma unroll
        for (int v = 0; v < 8; ++v) mx[v] = fmaxf(p0[v], p1[v]);
#pragma unroll
        for (int off = 1; off < 16; off <<= 1)
#pragma unroll
            for (int v = 0; v < 8; ++v)
                mx[v] = fmaxf(mx[v], __shfl_xor(mx[v], off, 32));

        float alpha[8];
#pragma unroll
        for (int v = 0; v < 8; ++v) {
            float mn = fmaxf(mrun[v], mx[v]);
            alpha[v] = __expf(mrun[v] - mn);
            mrun[v]  = mn;
            p0[v] = __expf(p0[v] - mn);
            p1[v] = __expf(p1[v] - mn);
        }
        float rs[8];
#pragma unroll
        for (int v = 0; v < 8; ++v) rs[v] = p0[v] + p1[v];
#pragma unroll
        for (int off = 1; off < 16; off <<= 1)
#pragma unroll
            for (int v = 0; v < 8; ++v)
                rs[v] += __shfl_xor(rs[v], off, 32);
#pragma unroll
        for (int v = 0; v < 8; ++v) lrun[v] = lrun[v] * alpha[v] + rs[v];
#pragma unroll
        for (int t = 0; t < 8; ++t)
#pragma unroll
            for (int v = 0; v < 8; ++v) acc[t][v] *= alpha[v];

        _Float16* lp = &ldsP[wv][0];
#pragma unroll
        for (int v = 0; v < 8; ++v) {
            lp[(khalf * 8 + v) * 32 + n]      = (_Float16)p0[v];
            lp[(khalf * 8 + v) * 32 + 16 + n] = (_Float16)p1[v];
        }
        asm volatile("s_wait_dscnt 0" ::: "memory");   // per-wave LDS RAW

        v16h pf = lds_frag(&ldsP[wv][0], n);

#pragma unroll
        for (int t = 0; t < 8; ++t) {
            v16h vf = load_frag(Vbase, S_, t * 16 + n, kb);
            acc[t] = wmma_f16(pf, vf, acc[t]);
        }
    }

#pragma unroll
    for (int t = 0; t < 8; ++t)
#pragma unroll
        for (int v = 0; v < 8; ++v) {
            float o = acc[t][v] / lrun[v];
            AO[((size_t)b * S_ + rbase + v) * (H_ * D_) +
               h * D_ + t * 16 + n] = (_Float16)o;
        }
}

// ---------------------------------------------------------------------------
// Host-side launch
// ---------------------------------------------------------------------------
extern "C" void kernel_launch(void* const* d_in, const int* in_sizes, int n_in,
                              void* d_out, int out_size, void* d_ws, size_t ws_size,
                              hipStream_t stream) {
    const float* hidden = (const float*)d_in[0];   // (B,S,DIM)
    const float* fcos   = (const float*)d_in[1];   // (S,64)
    const float* fsin   = (const float*)d_in[2];   // (S,64)
    // d_in[3]: mask (unused; causal mask synthesized in-kernel)
    const float* w_qkv  = (const float*)d_in[4];   // (4096,2048)
    const float* w_o    = (const float*)d_in[5];   // (2048,2048)
    const float* qnw    = (const float*)d_in[6];   // (128,)
    const float* knw    = (const float*)d_in[7];   // (128,)
    float* out = (float*)d_out;                    // (B,S,DIM) f32

    const size_t nBS = (size_t)B_ * S_;            // 4096
    char* ws = (char*)d_ws;
    size_t off = 0;
    auto take = [&](size_t bytes) -> void* {
        void* p = ws + off;
        off += (bytes + 255) & ~(size_t)255;
        return p;
    };
    _Float16* Xh    = (_Float16*)take(nBS * DIM_ * 2);
    _Float16* Wqkvh = (_Float16*)take((size_t)QKVC * DIM_ * 2);
    _Float16* Woh   = (_Float16*)take((size_t)DIM_ * (H_ * D_) * 2);
    float*    qkvf  = (float*)take(nBS * QKVC * 4);
    _Float16* Qh    = (_Float16*)take((size_t)B_ * H_  * S_ * D_ * 2);
    _Float16* Kh    = (_Float16*)take((size_t)B_ * KV_ * S_ * D_ * 2);
    _Float16* Vt    = (_Float16*)take((size_t)B_ * KV_ * D_ * S_ * 2);
    _Float16* AO    = (_Float16*)take(nBS * (size_t)(H_ * D_) * 2);
    (void)ws_size; (void)n_in; (void)in_sizes; (void)out_size;

    // 1) fp32 -> fp16 converts (4-wide)
    {
        int n1 = (int)(nBS * DIM_) / 4;
        int n2 = (QKVC * DIM_) / 4;
        int n3 = (DIM_ * H_ * D_) / 4;
        cvt_f32_f16<<<(n1 + 255) / 256, 256, 0, stream>>>(
            (const float4*)hidden, (v4h*)Xh, n1);
        cvt_f32_f16<<<(n2 + 255) / 256, 256, 0, stream>>>(
            (const float4*)w_qkv, (v4h*)Wqkvh, n2);
        cvt_f32_f16<<<(n3 + 255) / 256, 256, 0, stream>>>(
            (const float4*)w_o, (v4h*)Woh, n3);
    }

    // 2) QKV projection: qkv = X @ Wqkv^T   (M=4096, N=4096, K=2048)
    gemm_f16_wmma<QKVC, DIM_><<<dim3(QKVC / 128, (int)(nBS / 64)), 128, 0, stream>>>(
        Xh, Wqkvh, qkvf);

    // 3) RMSNorm + RoPE + layout shuffle (131072 waves, 8 waves/block)
    normrope_kernel<<<(B_ * S_ * 32) / 8, 256, 0, stream>>>(
        qkvf, qnw, knw, fcos, fsin, Qh, Kh, Vt);

    // 4) flash attention (4096 q-tiles, 4 waves/block)
    attn_kernel<<<(B_ * H_ * (S_ / 16)) / 4, 128, 0, stream>>>(Qh, Kh, Vt, AO);

    // 5) output projection: out = AO @ Wo^T  (M=4096, N=2048, K=2048)
    gemm_f16_wmma<DIM_, H_ * D_><<<dim3(DIM_ / 128, (int)(nBS / 64)), 128, 0, stream>>>(
        AO, Woh, out);
}
